// AttentionPairBias_30949534334980
// MI455X (gfx1250) — compile-verified, matrix operand order checked by reading, not run on hardware
//
#include <hip/hip_runtime.h>
#include <cstddef>

typedef __attribute__((ext_vector_type(16))) __bf16 v16bf;
typedef __attribute__((ext_vector_type(8)))  float  v8f;

#define NSEQ 1024
#define CS   768
#define CZ   128
#define NH   12
#define DH   64

union V16 {
    v16bf v;
    uint4  u[2];
    __bf16 e[16];
};

__device__ __forceinline__ v8f wmma_bf16(v16bf a, v16bf b, v8f c) {
    return __builtin_amdgcn_wmma_f32_16x16x32_bf16(false, a, false, b, (short)0, c, false, false);
}

// ---------------------------------------------------------------------------
// Pack weight matrix W[k][n] (f32, K x N) into transposed bf16 WT[n][k].
// ---------------------------------------------------------------------------
__global__ void pack_wt_kernel(const float* __restrict__ W, __bf16* __restrict__ WT,
                               int K, int N) {
    int idx = blockIdx.x * blockDim.x + threadIdx.x;
    if (idx >= K * N) return;
    int n = idx / K;
    int k = idx % K;
    WT[idx] = (__bf16)W[(size_t)k * N + n];
}

// ---------------------------------------------------------------------------
// Pre-build the WMMA B-layout fragment of Wz (c_z=128 -> 12 heads, padded 16)
// for each of the 4 K=32 steps and each of the 32 lanes: WzB[ks][lane][16].
// 4KB total, L2-resident; the 512MB-streaming zb kernel then loads its B
// operand with two b128s instead of 64 scalar f32 loads + converts per wave.
// ---------------------------------------------------------------------------
__global__ void pack_wzb_kernel(const float* __restrict__ Wz, __bf16* __restrict__ WzB) {
    int t = threadIdx.x;          // 128 threads: 4 ksteps x 32 lanes
    int ks = t >> 5, lane = t & 31;
    int half = lane >> 4, lr = lane & 15, kb = half * 8;
    int cb = ks * 32 + kb;
    __bf16* dst = WzB + (size_t)t * 16;
#pragma unroll
    for (int idx = 0; idx < 8; idx++) {
        int c0 = cb + idx, c1 = cb + 16 + idx;
        dst[idx]     = (lr < NH) ? (__bf16)Wz[c0 * NH + lr] : (__bf16)0.0f;
        dst[8 + idx] = (lr < NH) ? (__bf16)Wz[c1 * NH + lr] : (__bf16)0.0f;
    }
}

// ---------------------------------------------------------------------------
// LayerNorm over c_s=768, one block per row, output bf16.
// ---------------------------------------------------------------------------
__global__ void ln_s_kernel(const float* __restrict__ s, const float* __restrict__ sc,
                            const float* __restrict__ bi, __bf16* __restrict__ sn) {
    __shared__ float red1[256], red2[256];
    int row = blockIdx.x, tid = threadIdx.x;
    const float* x = s + (size_t)row * CS;
    float a = 0.f, b = 0.f, v[3];
#pragma unroll
    for (int t = 0; t < 3; t++) { float xv = x[tid + t * 256]; v[t] = xv; a += xv; b += xv * xv; }
    red1[tid] = a; red2[tid] = b; __syncthreads();
    for (int st = 128; st > 0; st >>= 1) {
        if (tid < st) { red1[tid] += red1[tid + st]; red2[tid] += red2[tid + st]; }
        __syncthreads();
    }
    float mu = red1[0] / CS;
    float var = red2[0] / CS - mu * mu;
    float rs = rsqrtf(var + 1e-5f);
#pragma unroll
    for (int t = 0; t < 3; t++) {
        int c = tid + t * 256;
        sn[(size_t)row * CS + c] = (__bf16)((v[t] - mu) * rs * sc[c] + bi[c]);
    }
}

// ---------------------------------------------------------------------------
// Generic WMMA GEMM: A[M x K] bf16 row-major  x  BT[N x K] bf16 (B transposed)
// One wave per 16x16 output tile, K-loop in steps of 32 (bf16 WMMA).
// mode 0: bf16 out row-major        mode 1: +bias, bf16 out
// mode 2: sigmoid, f32 out          mode 3: bf16 out transposed [n][m]
// mode 4: f32 out row-major
// ---------------------------------------------------------------------------
__global__ void gemm_wmma_kernel(const __bf16* __restrict__ A, const __bf16* __restrict__ BT,
                                 const float* __restrict__ bias, void* __restrict__ out,
                                 int M, int N, int K, int mode) {
    int lane = threadIdx.x & 31, wid = threadIdx.x >> 5;
    int ntiles = N / 16;
    int wg = blockIdx.x * (blockDim.x >> 5) + wid;
    if (wg >= (M / 16) * ntiles) return;
    int tm = wg / ntiles, tn = wg % ntiles;
    int half = lane >> 4, lr = lane & 15, kb = half * 8;

    const __bf16* arow = A  + (size_t)(tm * 16 + lr) * K;
    const __bf16* brow = BT + (size_t)(tn * 16 + lr) * K;
    v8f c = {};
    for (int k0 = 0; k0 < K; k0 += 32) {
        V16 a, b;
        a.u[0] = *(const uint4*)(arow + k0 + kb);
        a.u[1] = *(const uint4*)(arow + k0 + kb + 16);
        b.u[0] = *(const uint4*)(brow + k0 + kb);
        b.u[1] = *(const uint4*)(brow + k0 + kb + 16);
        c = wmma_bf16(a.v, b.v, c);
    }
    int n = tn * 16 + lr;
#pragma unroll
    for (int r = 0; r < 8; r++) {
        int m = tm * 16 + r + 8 * half;
        float val = c[r];
        if (mode == 1) val += bias[n];
        if (mode == 0 || mode == 1) {
            ((__bf16*)out)[(size_t)m * N + n] = (__bf16)val;
        } else if (mode == 2) {
            ((float*)out)[(size_t)m * N + n] = 1.0f / (1.0f + __expf(-val));
        } else if (mode == 3) {
            ((__bf16*)out)[(size_t)n * M + m] = (__bf16)val;
        } else {
            ((float*)out)[(size_t)m * N + n] = val;
        }
    }
}

// ---------------------------------------------------------------------------
// zb[h][i][j] = layernorm(z[i,j,:]) @ Wz   -- streams the 512MB z tensor once
// with explicit b128 loads. One wave handles (i, 16 consecutive j); lane L
// owns row j0+(L%16) with c-chunks per the WMMA bf16 A layout. 4 WMMAs
// contract c=128 against the precomputed WzB fragments.
// ---------------------------------------------------------------------------
__global__ void zb_kernel(const float* __restrict__ z, const float* __restrict__ zsc,
                          const float* __restrict__ zbi, const __bf16* __restrict__ WzB,
                          float* __restrict__ zb) {
    int lane = threadIdx.x & 31, wid = threadIdx.x >> 5;
    int w = blockIdx.x * 8 + wid;
    int i = w >> 6, jt = w & 63;
    int half = lane >> 4, lr = lane & 15, kb = half * 8;
    int j = jt * 16 + lr;
    const float* zr = z + ((size_t)i * NSEQ + j) * CZ;

    // 16 explicit b128 loads per lane: this lane's half-row (64 f32)
    float4 vv[16];
#pragma unroll
    for (int ks = 0; ks < 4; ks++) {
        int cb = ks * 32 + kb;
        vv[ks * 4 + 0] = *(const float4*)(zr + cb);
        vv[ks * 4 + 1] = *(const float4*)(zr + cb + 4);
        vv[ks * 4 + 2] = *(const float4*)(zr + cb + 16);
        vv[ks * 4 + 3] = *(const float4*)(zr + cb + 20);
    }
    float sum = 0.f, sq = 0.f;
#pragma unroll
    for (int t = 0; t < 16; t++) {
        sum += vv[t].x + vv[t].y + vv[t].z + vv[t].w;
        sq  += vv[t].x * vv[t].x + vv[t].y * vv[t].y + vv[t].z * vv[t].z + vv[t].w * vv[t].w;
    }
    sum += __shfl_xor(sum, 16, 32);   // combine the two half-row lanes
    sq  += __shfl_xor(sq, 16, 32);
    float mu = sum / CZ;
    float var = sq / CZ - mu * mu;
    float rs = rsqrtf(var + 1e-5f);

    v8f c = {};
#pragma unroll
    for (int ks = 0; ks < 4; ks++) {
        V16 a, b;
        int cb = ks * 32 + kb;
        const float* f = (const float*)&vv[ks * 4];
#pragma unroll
        for (int t = 0; t < 8; t++) {
            int c0 = cb + t, c1 = cb + 16 + t;
            a.e[t]     = (__bf16)((f[t]     - mu) * rs * zsc[c0] + zbi[c0]);
            a.e[8 + t] = (__bf16)((f[8 + t] - mu) * rs * zsc[c1] + zbi[c1]);
        }
        const uint4* bp = (const uint4*)(WzB + (size_t)(ks * 32 + lane) * 16);
        b.u[0] = bp[0];
        b.u[1] = bp[1];
        c = wmma_bf16(a.v, b.v, c);
    }
    if (lr < NH) {
#pragma unroll
        for (int r = 0; r < 8; r++) {
            int jl = r + 8 * half;
            zb[((size_t)lr * NSEQ + i) * NSEQ + jt * 16 + jl] = c[r];
        }
    }
}

// ---------------------------------------------------------------------------
// Flash-style attention: one wave per (head, 16-row i-tile). Streams K/V in
// j-tiles of 32, adds zb bias + mask, online softmax (shfl row reductions),
// P transposed C-layout -> A-layout through a per-wave LDS slice, PV via WMMA.
// Epilogue applies sigmoid gate g and stores bf16 og for the final GEMM.
// ---------------------------------------------------------------------------
__global__ void attn_kernel(const __bf16* __restrict__ q, const __bf16* __restrict__ k,
                            const __bf16* __restrict__ vT, const float* __restrict__ g,
                            const float* __restrict__ zb, const float* __restrict__ mask,
                            __bf16* __restrict__ og) {
    __shared__ __bf16 plds[8][16][32];
    int lane = threadIdx.x & 31, wid = threadIdx.x >> 5;
    int w = blockIdx.x * 8 + wid;            // 12 heads * 64 i-tiles = 768 waves
    int h = w >> 6, it = w & 63;
    int half = lane >> 4, lr = lane & 15, kb = half * 8;
    int i0 = it * 16;

    // Q tile in A layout for the two K=32 steps over d=64
    V16 aq[2];
    const __bf16* qrow = q + (size_t)(i0 + lr) * CS + h * DH;
    aq[0].u[0] = *(const uint4*)(qrow + kb);
    aq[0].u[1] = *(const uint4*)(qrow + kb + 16);
    aq[1].u[0] = *(const uint4*)(qrow + 32 + kb);
    aq[1].u[1] = *(const uint4*)(qrow + 32 + kb + 16);

    float mrun[8], lrun[8];
    v8f acc[4];
#pragma unroll
    for (int r = 0; r < 8; r++) { mrun[r] = -1e30f; lrun[r] = 0.f; }
#pragma unroll
    for (int dc = 0; dc < 4; dc++) acc[dc] = (v8f){};

    for (int jt = 0; jt < 32; jt++) {
        int j0 = jt * 32;
        v8f S[2];
#pragma unroll
        for (int jh = 0; jh < 2; jh++) {
            const __bf16* krow = k + (size_t)(j0 + jh * 16 + lr) * CS + h * DH;
            V16 b0, b1;
            b0.u[0] = *(const uint4*)(krow + kb);
            b0.u[1] = *(const uint4*)(krow + kb + 16);
            b1.u[0] = *(const uint4*)(krow + 32 + kb);
            b1.u[1] = *(const uint4*)(krow + 32 + kb + 16);
            v8f s = {};
            s = wmma_bf16(aq[0].v, b0.v, s);
            s = wmma_bf16(aq[1].v, b1.v, s);
            float mt = (1.0f - mask[j0 + jh * 16 + lr]) * -1000000.0f;
            const float* zrow = zb + ((size_t)h * NSEQ + i0) * NSEQ + j0 + jh * 16 + lr;
#pragma unroll
            for (int r = 0; r < 8; r++) {
                float zv = zrow[(size_t)(r + 8 * half) * NSEQ];
                s[r] = s[r] * 0.125f + zv + mt;   // 1/sqrt(64) = 0.125
            }
            S[jh] = s;
        }
        // online softmax, rows live across the 16 lanes of each half-wave
#pragma unroll
        for (int r = 0; r < 8; r++) {
            float tmax = fmaxf(S[0][r], S[1][r]);
            for (int off = 8; off > 0; off >>= 1) tmax = fmaxf(tmax, __shfl_xor(tmax, off, 32));
            float mn = fmaxf(mrun[r], tmax);
            float corr = __expf(mrun[r] - mn);
            float p0 = __expf(S[0][r] - mn);
            float p1 = __expf(S[1][r] - mn);
            float rsum = p0 + p1;
            for (int off = 8; off > 0; off >>= 1) rsum += __shfl_xor(rsum, off, 32);
            lrun[r] = lrun[r] * corr + rsum;
            mrun[r] = mn;
#pragma unroll
            for (int dc = 0; dc < 4; dc++) acc[dc][r] *= corr;
            S[0][r] = p0; S[1][r] = p1;
        }
        // transpose P (C layout) -> A layout via per-wave LDS slice
#pragma unroll
        for (int r = 0; r < 8; r++) {
            plds[wid][r + 8 * half][lr]      = (__bf16)S[0][r];
            plds[wid][r + 8 * half][16 + lr] = (__bf16)S[1][r];
        }
        V16 pa;
        pa.u[0] = *(const uint4*)(&plds[wid][lr][kb]);
        pa.u[1] = *(const uint4*)(&plds[wid][lr][kb + 16]);
        // PV: B = V[32 j x 16 d] from vT[d][j] (contiguous in j)
#pragma unroll
        for (int dc = 0; dc < 4; dc++) {
            const __bf16* vrow = vT + (size_t)(h * DH + dc * 16 + lr) * NSEQ + j0;
            V16 vb;
            vb.u[0] = *(const uint4*)(vrow + kb);
            vb.u[1] = *(const uint4*)(vrow + kb + 16);
            acc[dc] = wmma_bf16(pa.v, vb.v, acc[dc]);
        }
    }
    // epilogue: normalize, gate, store bf16
#pragma unroll
    for (int dc = 0; dc < 4; dc++) {
        int n = h * DH + dc * 16 + lr;
#pragma unroll
        for (int r = 0; r < 8; r++) {
            int m = i0 + r + 8 * half;
            float o = acc[dc][r] / lrun[r];
            og[(size_t)m * CS + n] = (__bf16)(o * g[(size_t)m * CS + n]);
        }
    }
}

// ---------------------------------------------------------------------------
// Host launcher
// ---------------------------------------------------------------------------
extern "C" void kernel_launch(void* const* d_in, const int* in_sizes, int n_in,
                              void* d_out, int out_size, void* d_ws, size_t ws_size,
                              hipStream_t stream) {
    (void)in_sizes; (void)n_in; (void)out_size; (void)ws_size;
    const float* s_in   = (const float*)d_in[0];
    const float* z_in   = (const float*)d_in[1];
    const float* mask   = (const float*)d_in[2];
    const float* ns_sc  = (const float*)d_in[3];
    const float* ns_bi  = (const float*)d_in[4];
    const float* Wq     = (const float*)d_in[5];
    const float* bq     = (const float*)d_in[6];
    const float* Wk     = (const float*)d_in[7];
    const float* Wv     = (const float*)d_in[8];
    const float* Wg     = (const float*)d_in[9];
    const float* nz_sc  = (const float*)d_in[10];
    const float* nz_bi  = (const float*)d_in[11];
    const float* Wz     = (const float*)d_in[12];
    const float* Wo     = (const float*)d_in[13];
    float* out = (float*)d_out;

    char* ws = (char*)d_ws;
    size_t off = 0;
    auto alloc = [&](size_t bytes) { size_t o = off; off += (bytes + 255) & ~(size_t)255; return o; };
    __bf16* sn   = (__bf16*)(ws + alloc((size_t)NSEQ * CS * 2));
    __bf16* WqT  = (__bf16*)(ws + alloc((size_t)CS * CS * 2));
    __bf16* WkT  = (__bf16*)(ws + alloc((size_t)CS * CS * 2));
    __bf16* WvT  = (__bf16*)(ws + alloc((size_t)CS * CS * 2));
    __bf16* WgT  = (__bf16*)(ws + alloc((size_t)CS * CS * 2));
    __bf16* WoT  = (__bf16*)(ws + alloc((size_t)CS * CS * 2));
    __bf16* qb   = (__bf16*)(ws + alloc((size_t)NSEQ * CS * 2));
    __bf16* kbuf = (__bf16*)(ws + alloc((size_t)NSEQ * CS * 2));
    __bf16* vT   = (__bf16*)(ws + alloc((size_t)CS * NSEQ * 2));
    float*  gbuf = (float*) (ws + alloc((size_t)NSEQ * CS * 4));
    __bf16* og   = (__bf16*)(ws + alloc((size_t)NSEQ * CS * 2));
    __bf16* WzB  = (__bf16*)(ws + alloc((size_t)4 * 32 * 16 * 2));
    float*  zb   = (float*) (ws + alloc((size_t)NH * NSEQ * NSEQ * 4));

    const int packBlocks = (CS * CS + 255) / 256;
    pack_wt_kernel<<<packBlocks, 256, 0, stream>>>(Wq, WqT, CS, CS);
    pack_wt_kernel<<<packBlocks, 256, 0, stream>>>(Wk, WkT, CS, CS);
    pack_wt_kernel<<<packBlocks, 256, 0, stream>>>(Wv, WvT, CS, CS);
    pack_wt_kernel<<<packBlocks, 256, 0, stream>>>(Wg, WgT, CS, CS);
    pack_wt_kernel<<<packBlocks, 256, 0, stream>>>(Wo, WoT, CS, CS);
    pack_wzb_kernel<<<1, 128, 0, stream>>>(Wz, WzB);

    ln_s_kernel<<<NSEQ, 256, 0, stream>>>(s_in, ns_sc, ns_bi, sn);

    const int gemmBlocks = (NSEQ / 16) * (CS / 16) / 8;   // 384
    gemm_wmma_kernel<<<gemmBlocks, 256, 0, stream>>>(sn, WqT, bq,      qb,   NSEQ, CS, CS, 1);
    gemm_wmma_kernel<<<gemmBlocks, 256, 0, stream>>>(sn, WkT, nullptr, kbuf, NSEQ, CS, CS, 0);
    gemm_wmma_kernel<<<gemmBlocks, 256, 0, stream>>>(sn, WvT, nullptr, vT,   NSEQ, CS, CS, 3);
    gemm_wmma_kernel<<<gemmBlocks, 256, 0, stream>>>(sn, WgT, nullptr, gbuf, NSEQ, CS, CS, 2);

    zb_kernel<<<NSEQ * 64 / 8, 256, 0, stream>>>(z_in, nz_sc, nz_bi, WzB, zb);

    attn_kernel<<<NH * 64 / 8, 256, 0, stream>>>(qb, kbuf, vT, gbuf, zb, mask, og);

    gemm_wmma_kernel<<<gemmBlocks, 256, 0, stream>>>(og, WoT, nullptr, out, NSEQ, CS, CS, 4);
}
